// CasualSelfAttention2_21827023799010
// MI455X (gfx1250) — compile-verified, matrix operand order checked by reading, not run on hardware
//
#include <hip/hip_runtime.h>
#include <hip/hip_bf16.h>

// ---------------------------------------------------------------- constants
constexpr int BB = 2, SS = 2048, EE = 4096;
constexpr int HH = 32, KVH = 8, DD = 128;
constexpr int KVE = KVH * DD;                 // 1024
constexpr float SM_SCALE = 0.08838834764831845f;  // 1/sqrt(128)

typedef __bf16 bf16;
typedef bf16  v16bf __attribute__((ext_vector_type(16)));
typedef bf16  v8bf  __attribute__((ext_vector_type(8)));
typedef float v8f   __attribute__((ext_vector_type(8)));
typedef unsigned int u32x4 __attribute__((ext_vector_type(4)));
typedef int i32x4 __attribute__((ext_vector_type(4)));
typedef int i32x8 __attribute__((ext_vector_type(8)));

union BF16Frag { v16bf v; v8bf h[2]; };

#if defined(__AMDGCN__) && __has_builtin(__builtin_amdgcn_tensor_load_to_lds) && \
    __has_builtin(__builtin_amdgcn_s_wait_tensorcnt)
#define HAS_TDM 1
#else
#define HAS_TDM 0
#endif

// ---------------------------------------------------------------- TDM: DMA a 64(rows) x 64(k) bf16
// tile from global (row stride = row_stride elems) into LDS at byte offset ldsoff,
// padding each 128B row with 16B so LDS row pitch = 144B (72 halves) to break bank conflicts.
__device__ __forceinline__ void tdm_issue_64x64(const bf16* gsrc, unsigned nrows,
                                                unsigned klen, unsigned row_stride,
                                                unsigned ldsoff)
{
#if HAS_TDM
    const unsigned long long ga = (unsigned long long)(uintptr_t)gsrc;
    u32x4 g0 = {
        1u,                                                  // count=1 (valid descriptor)
        ldsoff,                                              // lds_addr (bytes)
        (unsigned)(ga & 0xFFFFFFFFu),                        // global_addr[31:0]
        (unsigned)((ga >> 32) & 0x01FFFFFFu) | (2u << 30)    // global_addr[56:32] | type=2
    };
    // group1: data_size=1(2B) | pad_enable | pad_interval=4(32 DWORDs) | pad_amount=3(4 DWORDs)
    const unsigned w0 = (1u << 16) | (1u << 20) | (4u << 22) | (3u << 25);
    i32x8 g1 = {
        (int)w0,
        (int)((klen & 0xFFFFu) << 16),                       // tensor_dim0[15:0]
        (int)((klen >> 16) | ((nrows & 0xFFFFu) << 16)),     // tensor_dim0[31:16] | tensor_dim1[15:0]
        (int)((nrows >> 16) | (64u << 16)),                  // tensor_dim1[31:16] | tile_dim0=64
        (int)64,                                             // tile_dim1=64, tile_dim2=0
        (int)row_stride,                                     // tensor_dim0_stride[31:0]
        0, 0
    };
    i32x4 z4 = {0, 0, 0, 0};
#if defined(__clang_major__) && __clang_major__ >= 23
    i32x8 z8 = {0, 0, 0, 0, 0, 0, 0, 0};
    __builtin_amdgcn_tensor_load_to_lds(g0, g1, z4, z4, z8, 0);
#else
    __builtin_amdgcn_tensor_load_to_lds(g0, g1, z4, z4, 0);
#endif
#else
    (void)gsrc; (void)nrows; (void)klen; (void)row_stride; (void)ldsoff;
#endif
}

// ---------------------------------------------------------------- cast f32 -> bf16
__global__ void cast_f32_bf16(const float* __restrict__ in, bf16* __restrict__ out, size_t n) {
    size_t i = (size_t)blockIdx.x * blockDim.x + threadIdx.x;
    size_t stride = (size_t)gridDim.x * blockDim.x;
    for (; i < n; i += stride) out[i] = (bf16)in[i];
}

// ---------------------------------------------------------------- GEMM: Y[m,n] = sum_k A[m,k]*W[n,k]
// A: [M,K] bf16 row-major.  W: [N,K] bf16 row-major (computing A @ W^T).
// Block = 256 threads = 8 waves.  Workgroup tile 128(M) x 64(N); wave tile 16 x 64.
// Weight panel (64 x 64k) is TDM-DMA'd into double-buffered LDS; waves consume B from LDS.
template <bool F32OUT>
__global__ __launch_bounds__(256) void gemm_bf16_wmma(
    const bf16* __restrict__ A, const bf16* __restrict__ W,
    void* __restrict__ Yout, int M, int N, int K)
{
    const int lane  = threadIdx.x & 31;
    const int wave  = threadIdx.x >> 5;
    const int col   = lane & 15;      // A-row / W-row within tile, also C column
    const int khalf = lane >> 4;      // selects K sub-block of fragment
    const int mbase = blockIdx.y * 128 + wave * 16;
    const int nbase = blockIdx.x * 64;

    v8f acc[4] = {};
    const bf16* arow = A + (size_t)(mbase + col) * K;

#if HAS_TDM
    __shared__ __align__(16) bf16 bstage[2][64][72];   // 72-half pitch (TDM row pad)
    const bf16* wpanel = W + (size_t)nbase * K;

    if (wave == 0)
        tdm_issue_64x64(wpanel, (unsigned)N, (unsigned)K, (unsigned)K,
                        (unsigned)(uintptr_t)&bstage[0][0][0]);

    for (int k0 = 0; k0 < K; k0 += 64) {
        const int p = (k0 >> 6) & 1;
        if (wave == 0) __builtin_amdgcn_s_wait_tensorcnt(0);
        __syncthreads();
        if (wave == 0 && (k0 + 64) < K)
            tdm_issue_64x64(wpanel + (k0 + 64), (unsigned)N, (unsigned)K, (unsigned)K,
                            (unsigned)(uintptr_t)&bstage[p ^ 1][0][0]);
#pragma unroll
        for (int c = 0; c < 2; ++c) {
            const int ka = k0 + c * 32 + khalf * 8;
            BF16Frag a;
            a.h[0] = *(const v8bf*)(arow + ka);
            a.h[1] = *(const v8bf*)(arow + ka + 16);
            __builtin_prefetch(arow + ka + 64, 0, 1);
#pragma unroll
            for (int j = 0; j < 4; ++j) {
                BF16Frag b;
                const bf16* bp = &bstage[p][j * 16 + col][c * 32 + khalf * 8];
                b.h[0] = *(const v8bf*)(bp);
                b.h[1] = *(const v8bf*)(bp + 16);
                acc[j] = __builtin_amdgcn_wmma_f32_16x16x32_bf16(
                    false, a.v, false, b.v, (short)0, acc[j], false, false);
            }
        }
    }
#else
    // Fallback: B fragments straight from global.
    const bf16* wrow0 = W + (size_t)(nbase + col) * K;
    for (int k0 = 0; k0 < K; k0 += 32) {
        const int ka = k0 + khalf * 8;
        BF16Frag a;
        a.h[0] = *(const v8bf*)(arow + ka);
        a.h[1] = *(const v8bf*)(arow + ka + 16);
#pragma unroll
        for (int j = 0; j < 4; ++j) {
            BF16Frag b;
            const bf16* wp = wrow0 + (size_t)(j * 16) * K + ka;
            b.h[0] = *(const v8bf*)(wp);
            b.h[1] = *(const v8bf*)(wp + 16);
            acc[j] = __builtin_amdgcn_wmma_f32_16x16x32_bf16(
                false, a.v, false, b.v, (short)0, acc[j], false, false);
        }
    }
#endif

    const int rowoff = khalf * 8;
#pragma unroll
    for (int j = 0; j < 4; ++j) {
        const int n = nbase + j * 16 + col;
#pragma unroll
        for (int r = 0; r < 8; ++r) {
            const int m = mbase + rowoff + r;
            if constexpr (F32OUT)
                ((float*)Yout)[(size_t)m * N + n] = acc[j][r];
            else
                ((bf16*)Yout)[(size_t)m * N + n] = (bf16)acc[j][r];
        }
    }
}

// ---------------------------------------------------------------- RoPE + head transpose
// in:  [B, S, NH*D] bf16 ; out: [B*NH, S, D] bf16 (rotated halves)
__global__ void rope_transpose(const bf16* __restrict__ in, bf16* __restrict__ out,
                               const int* __restrict__ pos, int NH)
{
    const int j  = threadIdx.x;      // 0..63  (half of D)
    const int s  = blockIdx.x;
    const int bh = blockIdx.y;
    const int b  = bh / NH, h = bh % NH;

    const float p   = (float)pos[s];
    // inv_freq = ROPE_BASE^(-j/64);  log2(500000)/64 = 0.29580575958...
    const float inv = exp2f(-(float)j * 0.29580575958354873f);
    float sn, cs;
    __sincosf(p * inv, &sn, &cs);

    const bf16* src = in + ((size_t)b * SS + s) * ((size_t)NH * DD) + (size_t)h * DD;
    bf16* dst = out + (((size_t)bh) * SS + s) * DD;
    const float x1 = (float)src[j];
    const float x2 = (float)src[j + 64];
    dst[j]      = (bf16)(x1 * cs - x2 * sn);
    dst[j + 64] = (bf16)(x2 * cs + x1 * sn);
}

// ---------------------------------------------------------------- V transpose: [B,S,KV*D] -> [B*KV, D, S]
__global__ void v_transpose(const bf16* __restrict__ in, bf16* __restrict__ out)
{
    const int d  = threadIdx.x;      // 0..127
    const int s  = blockIdx.x;
    const int bk = blockIdx.y;       // b*KV + kv
    const int b  = bk / KVH, kv = bk % KVH;
    out[((size_t)bk * DD + d) * SS + s] =
        in[((size_t)b * SS + s) * KVE + (size_t)kv * DD + d];
}

// ---------------------------------------------------------------- flash attention (causal, GQA)
// Qr: [B*H, S, D]  Kr: [B*KV, S, D]  Vt: [B*KV, D, S]  AO: [B, S, H*D] (bf16)
__global__ __launch_bounds__(256) void attn_wmma(
    const bf16* __restrict__ Qr, const bf16* __restrict__ Kr,
    const bf16* __restrict__ Vt, bf16* __restrict__ AO)
{
    __shared__ __align__(16) bf16 plds[8][16][32];   // per-wave P staging (1KB/wave)

    const int lane  = threadIdx.x & 31;
    const int wave  = threadIdx.x >> 5;
    const int col   = lane & 15;
    const int khalf = lane >> 4;
    const int bh = blockIdx.y;                       // b*H + h
    const int b  = bh / HH, h = bh % HH;
    const int kv = h / (HH / KVH);
    const int qb = blockIdx.x * 128 + wave * 16;     // this wave's q-row base

    const bf16* Qbase = Qr + ((size_t)bh * SS + qb) * DD;
    const bf16* Kbase = Kr + (size_t)(b * KVH + kv) * SS * DD;
    const bf16* Vbase = Vt + (size_t)(b * KVH + kv) * DD * SS;

    // Preload Q fragments (4 x 32-wide d chunks)
    BF16Frag qf[4];
    {
        const bf16* qrow = Qbase + (size_t)col * DD;
#pragma unroll
        for (int c = 0; c < 4; ++c) {
            const int d0 = c * 32 + khalf * 8;
            qf[c].h[0] = *(const v8bf*)(qrow + d0);
            qf[c].h[1] = *(const v8bf*)(qrow + d0 + 16);
        }
    }

    v8f acco[8] = {};
    float mrow[8], srow[8];
#pragma unroll
    for (int r = 0; r < 8; ++r) { mrow[r] = -1e30f; srow[r] = 0.0f; }

    const int nchunks = (qb + 15) / 32 + 1;          // causal: stop past the diagonal

    for (int kc = 0; kc < nchunks; ++kc) {
        const int kbase = kc * 32;

        // ---- scores: Q(16x128) x K^T -> two 16x16 tiles
        v8f sc[2] = {};
#pragma unroll
        for (int t = 0; t < 2; ++t) {
            const bf16* krow = Kbase + (size_t)(kbase + t * 16 + col) * DD;
#pragma unroll
            for (int c = 0; c < 4; ++c) {
                BF16Frag kb;
                const int d0 = c * 32 + khalf * 8;
                kb.h[0] = *(const v8bf*)(krow + d0);
                kb.h[1] = *(const v8bf*)(krow + d0 + 16);
                sc[t] = __builtin_amdgcn_wmma_f32_16x16x32_bf16(
                    false, qf[c].v, false, kb.v, (short)0, sc[t], false, false);
            }
        }

        // ---- scale + causal mask
        const bool edge = (kbase + 31 > qb);
#pragma unroll
        for (int t = 0; t < 2; ++t) {
            const int key = kbase + t * 16 + col;
#pragma unroll
            for (int r = 0; r < 8; ++r) {
                float s = sc[t][r] * SM_SCALE;
                if (edge && key > qb + khalf * 8 + r) s = -1e30f;
                sc[t][r] = s;
            }
        }

        // ---- row max across the 16 lanes of each half
        float rowm[8];
#pragma unroll
        for (int r = 0; r < 8; ++r) rowm[r] = fmaxf(sc[0][r], sc[1][r]);
#pragma unroll
        for (int m = 1; m <= 8; m <<= 1)
#pragma unroll
            for (int r = 0; r < 8; ++r)
                rowm[r] = fmaxf(rowm[r], __shfl_xor(rowm[r], m, 32));

        // ---- online softmax update
        float alpha[8];
#pragma unroll
        for (int r = 0; r < 8; ++r) {
            const float mnew = fmaxf(mrow[r], rowm[r]);
            alpha[r] = __expf(mrow[r] - mnew);
            mrow[r]  = mnew;
        }
        float rsum[8];
#pragma unroll
        for (int r = 0; r < 8; ++r) rsum[r] = 0.0f;
#pragma unroll
        for (int t = 0; t < 2; ++t)
#pragma unroll
            for (int r = 0; r < 8; ++r) {
                const float p = __expf(sc[t][r] - mrow[r]);
                sc[t][r] = p;
                rsum[r] += p;
            }
#pragma unroll
        for (int m = 1; m <= 8; m <<= 1)
#pragma unroll
            for (int r = 0; r < 8; ++r)
                rsum[r] += __shfl_xor(rsum[r], m, 32);
#pragma unroll
        for (int r = 0; r < 8; ++r) srow[r] = srow[r] * alpha[r] + rsum[r];

        // ---- rescale running output
#pragma unroll
        for (int j = 0; j < 8; ++j)
#pragma unroll
            for (int r = 0; r < 8; ++r)
                acco[j][r] *= alpha[r];

        // ---- P: C/D layout -> A-fragment layout via per-wave LDS (in-order DS ops)
#pragma unroll
        for (int t = 0; t < 2; ++t)
#pragma unroll
            for (int r = 0; r < 8; ++r)
                plds[wave][khalf * 8 + r][t * 16 + col] = (bf16)sc[t][r];
        BF16Frag pf;
        pf.h[0] = *(const v8bf*)(&plds[wave][col][khalf * 8]);
        pf.h[1] = *(const v8bf*)(&plds[wave][col][16 + khalf * 8]);

        // ---- PV: out(16x128) += P(16x32) x V(32x128)
#pragma unroll
        for (int j = 0; j < 8; ++j) {
            BF16Frag vf;
            const bf16* vrow = Vbase + (size_t)(j * 16 + col) * SS + kbase;
            const int kk = khalf * 8;
            vf.h[0] = *(const v8bf*)(vrow + kk);
            vf.h[1] = *(const v8bf*)(vrow + kk + 16);
            acco[j] = __builtin_amdgcn_wmma_f32_16x16x32_bf16(
                false, pf.v, false, vf.v, (short)0, acco[j], false, false);
        }
    }

    // ---- epilogue: normalize and store to [B,S,H*D]
#pragma unroll
    for (int r = 0; r < 8; ++r) {
        const int q = qb + khalf * 8 + r;
        const float inv = 1.0f / srow[r];
        bf16* orow = AO + ((size_t)b * SS + q) * EE + (size_t)h * DD;
#pragma unroll
        for (int j = 0; j < 8; ++j)
            orow[j * 16 + col] = (bf16)(acco[j][r] * inv);
    }
}

// ---------------------------------------------------------------- launcher
extern "C" void kernel_launch(void* const* d_in, const int* in_sizes, int n_in,
                              void* d_out, int out_size, void* d_ws, size_t ws_size,
                              hipStream_t stream) {
    (void)in_sizes; (void)n_in; (void)out_size; (void)ws_size;
    const float* x  = (const float*)d_in[0];
    const float* wq = (const float*)d_in[1];
    const float* wk = (const float*)d_in[2];
    const float* wv = (const float*)d_in[3];
    const float* wo = (const float*)d_in[4];
    const int*  pos = (const int*)d_in[5];
    float* out = (float*)d_out;

    constexpr size_t NX  = (size_t)BB * SS * EE;    // 16.7M
    constexpr size_t NWQ = (size_t)EE * EE;
    constexpr size_t NWK = (size_t)KVE * EE;
    constexpr size_t NKV = (size_t)BB * SS * KVE;

    char* ws = (char*)d_ws;
    size_t off = 0;
    auto alloc = [&](size_t bytes) -> void* {
        void* p = ws + off;
        off += (bytes + 255) & ~(size_t)255;
        return p;
    };

    bf16* xb  = (bf16*)alloc(NX  * 2);
    bf16* wqb = (bf16*)alloc(NWQ * 2);
    bf16* wkb = (bf16*)alloc(NWK * 2);
    bf16* wvb = (bf16*)alloc(NWK * 2);
    bf16* wob = (bf16*)alloc(NWQ * 2);
    bf16* Qp  = (bf16*)alloc(NX  * 2);
    bf16* Kp  = (bf16*)alloc(NKV * 2);
    bf16* Vp  = (bf16*)alloc(NKV * 2);
    bf16* Qr  = (bf16*)alloc(NX  * 2);
    bf16* Kr  = (bf16*)alloc(NKV * 2);
    bf16* Vt  = (bf16*)alloc(NKV * 2);
    bf16* AO  = xb;   // x is dead after the projections; reuse its buffer

    // 1) casts
    cast_f32_bf16<<<8192, 256, 0, stream>>>(x,  xb,  NX);
    cast_f32_bf16<<<8192, 256, 0, stream>>>(wq, wqb, NWQ);
    cast_f32_bf16<<<4096, 256, 0, stream>>>(wk, wkb, NWK);
    cast_f32_bf16<<<4096, 256, 0, stream>>>(wv, wvb, NWK);
    cast_f32_bf16<<<8192, 256, 0, stream>>>(wo, wob, NWQ);

    // 2) projections: [4096, E] x W^T
    const int M = BB * SS;   // 4096
    gemm_bf16_wmma<false><<<dim3(EE / 64,  M / 128), 256, 0, stream>>>(xb, wqb, Qp, M, EE,  EE);
    gemm_bf16_wmma<false><<<dim3(KVE / 64, M / 128), 256, 0, stream>>>(xb, wkb, Kp, M, KVE, EE);
    gemm_bf16_wmma<false><<<dim3(KVE / 64, M / 128), 256, 0, stream>>>(xb, wvb, Vp, M, KVE, EE);

    // 3) RoPE + transpose
    rope_transpose<<<dim3(SS, BB * HH),  64, 0, stream>>>(Qp, Qr, pos, HH);
    rope_transpose<<<dim3(SS, BB * KVH), 64, 0, stream>>>(Kp, Kr, pos, KVH);
    v_transpose  <<<dim3(SS, BB * KVH), 128, 0, stream>>>(Vp, Vt);

    // 4) attention
    attn_wmma<<<dim3(SS / 128, BB * HH), 256, 0, stream>>>(Qr, Kr, Vt, AO);

    // 5) output projection (f32 out)
    gemm_bf16_wmma<true><<<dim3(EE / 64, M / 128), 256, 0, stream>>>(AO, wob, out, M, EE, EE);
}
